// CRSDSequence_71339406786970
// MI455X (gfx1250) — matrix operand order repaired and checked by reading, not compiled, not run on hardware
//
#include <hip/hip_runtime.h>
#include <math.h>

typedef __attribute__((ext_vector_type(16))) __bf16 v16bf;
typedef __attribute__((ext_vector_type(8)))  __bf16 v8bf;
typedef __attribute__((ext_vector_type(8)))  float  v8f;
typedef int v4i __attribute__((vector_size(16)));

constexpr int kB = 8;
constexpr int kT = 512;
constexpr int kE = 1024;
constexpr int kH = 1024;
constexpr int kD = 512;
constexpr int kV = 32000;
constexpr int kM = kB * kT;     // 4096 flattened (t,b) rows
constexpr int kNWG = 32;        // persistent workgroups in scan
constexpr int kGM = 64;         // GEMM block M rows (4 M-tiles per wave)
constexpr float kAlpha = 0.5f;

// ---------- async LDS staging (CDNA5 GLOBAL_LOAD_ASYNC_TO_LDS_B128) ----------
__device__ __forceinline__ void copy16_async(void* lds_dst, const void* gsrc) {
#if __has_builtin(__builtin_amdgcn_global_load_async_to_lds_b128)
  __builtin_amdgcn_global_load_async_to_lds_b128(
      (__attribute__((address_space(1))) v4i*)gsrc,
      (__attribute__((address_space(3))) v4i*)lds_dst, 0, 0);
#else
  *(uint4*)lds_dst = *(const uint4*)gsrc;
#endif
}
__device__ __forceinline__ void wait_async_copies() {
#if __has_builtin(__builtin_amdgcn_global_load_async_to_lds_b128)
#if __has_builtin(__builtin_amdgcn_s_wait_asynccnt)
  __builtin_amdgcn_s_wait_asynccnt(0);
#else
  asm volatile("s_wait_asynccnt 0x0" ::: "memory");
#endif
#endif
}

// ---------- WMMA fragment helpers (CDNA5 16x16x32 bf16 layouts) ----------
// A (16xK bf16): lanes 0-15 -> M=lane, halves = K kb+0..7 / kb+16..23
//                lanes 16-31 -> M=lane-16, halves = K kb+8..15 / kb+24..31
__device__ __forceinline__ v16bf load_a_frag(const __bf16* S, int pitch, int kb) {
  const int lane = threadIdx.x & 31;
  const int m = lane & 15;
  const int hi = lane >> 4;
  const __bf16* p = S + (size_t)m * pitch + kb + hi * 8;
  v8bf x = *(const v8bf*)(p);
  v8bf y = *(const v8bf*)(p + 16);
  return __builtin_shufflevector(x, y, 0,1,2,3,4,5,6,7,8,9,10,11,12,13,14,15);
}

// B (Kx16, from row-major W[N][K]): lane n holds column N=n, K kb+0..7 / kb+16..23
// (lanes 16-31: K kb+8..15 / kb+24..31) -> two 16-byte chunks from W row n.
__device__ __forceinline__ v16bf load_b_frag(const __bf16* Wt, int pitch, int kb) {
  const int lane = threadIdx.x & 31;
  const int n = lane & 15;
  const int hi = lane >> 4;
  const __bf16* p = Wt + (size_t)n * pitch + kb + hi * 8;
  v8bf x = *(const v8bf*)(p);
  v8bf y = *(const v8bf*)(p + 16);
  return __builtin_shufflevector(x, y, 0,1,2,3,4,5,6,7,8,9,10,11,12,13,14,15);
}

__device__ __forceinline__ v8f wmma_bf16(v16bf a, v16bf b, v8f c) {
  return __builtin_amdgcn_wmma_f32_16x16x32_bf16(false, a, false, b, (short)0, c,
                                                 false, false);
}

// ---------- small utility kernels ----------
__global__ void k_f2bf(const float* __restrict__ s, __bf16* __restrict__ d, int n) {
  for (int i = blockIdx.x * blockDim.x + threadIdx.x; i < n;
       i += gridDim.x * blockDim.x)
    d[i] = (__bf16)s[i];
}

// X[t*8+b][e] = emb[ids[b][t]][e]  (bf16)
__global__ void k_gather(const int* __restrict__ ids, const float* __restrict__ emb,
                         __bf16* __restrict__ X) {
  const int row = blockIdx.x;     // t*8 + b
  const int t = row >> 3;
  const int b = row & 7;
  const int id = ids[b * kT + t];
  const float* src = emb + (size_t)id * kE;
  __bf16* dst = X + (size_t)row * kE;
  for (int k = threadIdx.x; k < kE; k += blockDim.x) dst[k] = (__bf16)src[k];
}

__global__ void k_init(__bf16* __restrict__ S, int* __restrict__ bar) {
  int i = blockIdx.x * blockDim.x + threadIdx.x;
  if (i < 2 * 16 * 2048) S[i] = (__bf16)0.0f;   // both state parities -> 0
  if (i == 0) bar[0] = 0;                        // grid barrier counter
}

// ---------- batched WMMA GEMM:  C[4096][N] = A[4096][1024] * W[N][1024]^T + bias
// block: 256 thr = 8 waves; block tile = 64 rows x 128 cols; K = 1024 fixed.
// Each wave: 4 M-tiles x 1 N-tile -> each B-fragment feeds 4 WMMAs (4x less
// L2 traffic on the 64MB out_W than 1 tile/wave).
__global__ void k_gemm_bf16(const __bf16* __restrict__ A,
                            const __bf16* __restrict__ W,
                            const float* __restrict__ bias,
                            float* __restrict__ C, int N, int ngroups) {
  extern __shared__ __bf16 ldsA[];               // 64 x 1032 (padded pitch)
  const int mt = blockIdx.x / ngroups;
  const int ng = blockIdx.x % ngroups;
  const int m0 = mt * kGM;
  // stage A tile: 64 rows x 1024 bf16, async DMA into LDS
  for (int c = threadIdx.x; c < kGM * 128; c += blockDim.x) {
    const int r = c >> 7, k8 = c & 127;
    copy16_async(ldsA + r * 1032 + k8 * 8,
                 A + (size_t)(m0 + r) * 1024 + k8 * 8);
  }
  wait_async_copies();
  __syncthreads();
  const int wv = threadIdx.x >> 5;
  const int lane = threadIdx.x & 31;
  const int n0 = (ng * 8 + wv) * 16;
  const __bf16* Wt = W + (size_t)n0 * 1024;
  v8f acc[4] = {{}, {}, {}, {}};
  for (int ks = 0; ks < 32; ++ks) {
    const int kb = ks * 32;
    if (ks + 1 < 32)   // pull next B chunk toward near caches
      __builtin_prefetch(Wt + (size_t)(lane & 15) * 1024 + kb + 32, 0, 3);
    v16bf b = load_b_frag(Wt, 1024, kb);
    v16bf a0 = load_a_frag(ldsA + 0 * 16 * 1032, 1032, kb);
    v16bf a1 = load_a_frag(ldsA + 1 * 16 * 1032, 1032, kb);
    v16bf a2 = load_a_frag(ldsA + 2 * 16 * 1032, 1032, kb);
    v16bf a3 = load_a_frag(ldsA + 3 * 16 * 1032, 1032, kb);
    acc[0] = wmma_bf16(a0, b, acc[0]);
    acc[1] = wmma_bf16(a1, b, acc[1]);
    acc[2] = wmma_bf16(a2, b, acc[2]);
    acc[3] = wmma_bf16(a3, b, acc[3]);
  }
  const int n = n0 + (lane & 15);
  const float bv = bias ? bias[n] : 0.0f;
  const int mh = (lane >> 4) * 8;    // C VGPR r: lanes<16 -> M=r, lanes>=16 -> M=r+8
#pragma unroll
  for (int i = 0; i < 4; ++i)
#pragma unroll
    for (int r = 0; r < 8; ++r)
      C[(size_t)(m0 + i * 16 + mh + r) * N + n] = acc[i][r] + bv;
}

// ---------- grid barrier (monotonic counter, release/acquire fences) ----------
__device__ __forceinline__ void grid_barrier(int* bar, int target) {
  __threadfence();
  __syncthreads();
  if (threadIdx.x == 0) {
    atomicAdd(bar, 1);
    while (__atomic_load_n(bar, __ATOMIC_RELAXED) < target)
      __builtin_amdgcn_s_sleep(2);
  }
  __syncthreads();
  __threadfence();
}

// ---------- sequential scan: 32 persistent WGs, column-sharded state ----------
// State Sbuf[2][16][2048] bf16: cols [0,1024)=h, [1024,1536)=r0, [1536,2048)=r1.
// Rows 8..15 stay zero (WMMA M-padding). WG w owns h cols [w*32,+32),
// r cols [w*16,+16) per reservoir. W_h/U0/U1 slices LDS-resident.
__global__ void __launch_bounds__(256)
k_scan(const __bf16* __restrict__ Whbf, const __bf16* __restrict__ U0bf,
       const __bf16* __restrict__ U1bf, const __bf16* __restrict__ Wres0bf,
       const __bf16* __restrict__ Wres1bf, const float* __restrict__ XprojH,
       const float* __restrict__ XprojR0, const float* __restrict__ XprojR1,
       __bf16* __restrict__ Sbuf, __bf16* __restrict__ Hs, int* __restrict__ bar) {
  extern __shared__ char smem[];
  __bf16* Wh_l = (__bf16*)smem;                  // 32 x 1032 (padded)
  __bf16* U0_l = Wh_l + 32 * 1032;               // 32 x 520
  __bf16* U1_l = U0_l + 32 * 520;
  float* red = (float*)(smem + (size_t)(32 * 1032 + 2 * 32 * 520) * sizeof(__bf16));

  const int w = blockIdx.x;
  const int lane = threadIdx.x & 31;
  const int wv = threadIdx.x >> 5;

  // one-time async LDS preload of this WG's weight slices
  for (int c = threadIdx.x; c < 32 * 128; c += blockDim.x) {
    const int r = c >> 7, k8 = c & 127;
    copy16_async(Wh_l + r * 1032 + k8 * 8,
                 Whbf + (size_t)(w * 32 + r) * 1024 + k8 * 8);
  }
  for (int c = threadIdx.x; c < 32 * 64; c += blockDim.x) {
    const int r = c >> 6, k8 = c & 63;
    copy16_async(U0_l + r * 520 + k8 * 8,
                 U0bf + (size_t)(w * 32 + r) * 512 + k8 * 8);
    copy16_async(U1_l + r * 520 + k8 * 8,
                 U1bf + (size_t)(w * 32 + r) * 512 + k8 * 8);
  }
  wait_async_copies();
  __syncthreads();

  for (int t = 0; t < kT; ++t) {
    const int p = t & 1;
    const __bf16* Sold = Sbuf + (size_t)p * 16 * 2048;
    __bf16* Snew = Sbuf + (size_t)(1 - p) * 16 * 2048;

    // ---- Phase A: reservoirs. waves 0-3 -> r0, 4-7 -> r1; 4-way K split ----
    {
      const int res = wv >> 2;
      const int kq = wv & 3;
      const __bf16* Wt = (res ? Wres1bf : Wres0bf) + (size_t)(w * 16) * 512;
      v8f acc = {};
#pragma unroll
      for (int kk = 0; kk < 4; ++kk) {
        const int kb = (kq * 4 + kk) * 32;
        v16bf a = load_a_frag(Sold + 1024 + res * 512, 2048, kb);
        v16bf b = load_b_frag(Wt, 512, kb);
        acc = wmma_bf16(a, b, acc);
      }
      if (kq != 0) {
        float* rp = red + (wv * 32 + lane) * 8;
#pragma unroll
        for (int j = 0; j < 8; ++j) rp[j] = acc[j];
      }
      __syncthreads();
      if (kq == 0 && lane < 16) {
#pragma unroll
        for (int i = 1; i < 4; ++i) {
          const float* q = red + ((wv + i) * 32 + lane) * 8;
#pragma unroll
          for (int j = 0; j < 8; ++j) acc[j] += q[j];
        }
        const int col = w * 16 + lane;
        const float* Xp = res ? XprojR1 : XprojR0;
#pragma unroll
        for (int m = 0; m < 8; ++m) {
          const float pre = acc[m] + Xp[(size_t)(t * 8 + m) * 512 + col];
          const float v = tanhf(pre);
          const size_t idx = (size_t)m * 2048 + 1024 + res * 512 + col;
          const float ro = (float)Sold[idx];
          Snew[idx] = (__bf16)((1.0f - kAlpha) * ro + kAlpha * v);
        }
      }
    }
    grid_barrier(bar, (t * 2 + 1) * kNWG);   // all r_new slices visible

    // ---- Phase B: hidden. waves 0-3 -> cols nt=0, 4-7 -> nt=1; 4-way K split
    // K order: h(old, 0..1023) | r0(new, 1024..1535) | r1(new, 1536..2047) ----
    {
      const int nt = wv >> 2;
      const int kq = wv & 3;
      v8f acc = {};
      for (int kk = 0; kk < 16; ++kk) {
        const int kcol = (kq * 16 + kk) * 32;
        const __bf16* Ab = (kcol < 1024) ? Sold : Snew;
        v16bf a = load_a_frag(Ab, 2048, kcol);
        v16bf b;
        if (kcol < 1024)
          b = load_b_frag(Wh_l + (size_t)(nt * 16) * 1032, 1032, kcol);
        else if (kcol < 1536)
          b = load_b_frag(U0_l + (size_t)(nt * 16) * 520, 520, kcol - 1024);
        else
          b = load_b_frag(U1_l + (size_t)(nt * 16) * 520, 520, kcol - 1536);
        acc = wmma_bf16(a, b, acc);
      }
      if (kq != 0) {
        float* rp = red + (wv * 32 + lane) * 8;
#pragma unroll
        for (int j = 0; j < 8; ++j) rp[j] = acc[j];
      }
      __syncthreads();
      if (kq == 0 && lane < 16) {
#pragma unroll
        for (int i = 1; i < 4; ++i) {
          const float* q = red + ((wv + i) * 32 + lane) * 8;
#pragma unroll
          for (int j = 0; j < 8; ++j) acc[j] += q[j];
        }
        const int col = w * 32 + nt * 16 + lane;
#pragma unroll
        for (int m = 0; m < 8; ++m) {
          const float pre = acc[m] + XprojH[(size_t)(t * 8 + m) * 1024 + col];
          const float h = tanhf(pre);
          Snew[(size_t)m * 2048 + col] = (__bf16)h;
          Hs[(size_t)(m * kT + t) * 1024 + col] = (__bf16)h;  // row = b*T+t
        }
      }
    }
    grid_barrier(bar, (t * 2 + 2) * kNWG);   // h_new visible for next step
  }
}

// ---------- host orchestration ----------
extern "C" void kernel_launch(void* const* d_in, const int* in_sizes, int n_in,
                              void* d_out, int out_size, void* d_ws, size_t ws_size,
                              hipStream_t stream) {
  (void)in_sizes; (void)n_in; (void)out_size; (void)ws_size;
  const int*   ids   = (const int*)d_in[0];
  const float* emb   = (const float*)d_in[1];
  const float* W_x   = (const float*)d_in[2];
  const float* W_h   = (const float*)d_in[3];
  const float* b_h   = (const float*)d_in[4];
  const float* Win0  = (const float*)d_in[5];
  const float* Wres0 = (const float*)d_in[6];
  const float* U0    = (const float*)d_in[7];
  const float* Win1  = (const float*)d_in[8];
  const float* Wres1 = (const float*)d_in[9];
  const float* U1    = (const float*)d_in[10];
  const float* outW  = (const float*)d_in[11];
  const float* outb  = (const float*)d_in[12];
  float* out = (float*)d_out;

  char* ws = (char*)d_ws;
  size_t off = 0;
  auto alloc = [&](size_t bytes) -> void* {
    void* p = (void*)(ws + off);
    off = (off + bytes + 255) & ~(size_t)255;
    return p;
  };
  __bf16* Xbf     = (__bf16*)alloc((size_t)kM * kE * 2);
  __bf16* Wxbf    = (__bf16*)alloc((size_t)kH * kE * 2);
  __bf16* Whbf    = (__bf16*)alloc((size_t)kH * kH * 2);
  __bf16* Win0bf  = (__bf16*)alloc((size_t)kD * kE * 2);
  __bf16* Win1bf  = (__bf16*)alloc((size_t)kD * kE * 2);
  __bf16* U0bf    = (__bf16*)alloc((size_t)kH * kD * 2);
  __bf16* U1bf    = (__bf16*)alloc((size_t)kH * kD * 2);
  __bf16* Wres0bf = (__bf16*)alloc((size_t)kD * kD * 2);
  __bf16* Wres1bf = (__bf16*)alloc((size_t)kD * kD * 2);
  __bf16* OutWbf  = (__bf16*)alloc((size_t)kV * kE * 2);
  float*  XprojH  = (float*)alloc((size_t)kM * kH * 4);
  float*  XprojR0 = (float*)alloc((size_t)kM * kD * 4);
  float*  XprojR1 = (float*)alloc((size_t)kM * kD * 4);
  __bf16* Hsbf    = (__bf16*)alloc((size_t)kM * kH * 2);
  __bf16* Sbuf    = (__bf16*)alloc((size_t)2 * 16 * 2048 * 2);
  int*    bar     = (int*)alloc(256);

  k_init<<<256, 256, 0, stream>>>(Sbuf, bar);

  auto cvt = [&](const float* s, __bf16* d, int n) {
    int blocks = (n + 255) / 256;
    if (blocks > 4096) blocks = 4096;
    k_f2bf<<<blocks, 256, 0, stream>>>(s, d, n);
  };
  cvt(W_x, Wxbf, kH * kE);
  cvt(W_h, Whbf, kH * kH);
  cvt(Win0, Win0bf, kD * kE);
  cvt(Win1, Win1bf, kD * kE);
  cvt(U0, U0bf, kH * kD);
  cvt(U1, U1bf, kH * kD);
  cvt(Wres0, Wres0bf, kD * kD);
  cvt(Wres1, Wres1bf, kD * kD);
  cvt(outW, OutWbf, kV * kE);

  k_gather<<<kM, 256, 0, stream>>>(ids, emb, Xbf);

  const size_t ldsGemm = (size_t)kGM * 1032 * 2;   // 132096 B
  // input projections (hoisted out of the scan)
  k_gemm_bf16<<<(kM / kGM) * (kH / 128), 256, ldsGemm, stream>>>(
      Xbf, Wxbf, b_h, XprojH, kH, kH / 128);
  k_gemm_bf16<<<(kM / kGM) * (kD / 128), 256, ldsGemm, stream>>>(
      Xbf, Win0bf, nullptr, XprojR0, kD, kD / 128);
  k_gemm_bf16<<<(kM / kGM) * (kD / 128), 256, ldsGemm, stream>>>(
      Xbf, Win1bf, nullptr, XprojR1, kD, kD / 128);

  const size_t ldsScan =
      (size_t)(32 * 1032 + 2 * 32 * 520) * 2 + (size_t)8 * 32 * 8 * 4;  // 140800 B
  k_scan<<<kNWG, 256, ldsScan, stream>>>(Whbf, U0bf, U1bf, Wres0bf, Wres1bf,
                                         XprojH, XprojR0, XprojR1, Sbuf, Hsbf, bar);

  // output projection: logits[b*T+t][V] = Hs @ outW^T + out_b
  k_gemm_bf16<<<(kM / kGM) * (kV / 128), 256, ldsGemm, stream>>>(
      Hsbf, OutWbf, outb, out, kV, kV / 128);
}
